// CrossAttentionBlock_61761629716508
// MI455X (gfx1250) — matrix operand order repaired and checked by reading, not compiled, not run on hardware
//
#include <hip/hip_runtime.h>
#include <hip/hip_bf16.h>

#define NB 8
#define NT 1024
#define NC 768
#define NH 12
#define ND 64
#define BH (NB * NH)          // 96
#define SCALE 0.125f          // 64^-0.5

typedef __attribute__((ext_vector_type(16))) __bf16 v16bf;
typedef __attribute__((ext_vector_type(8)))  float  v8f;

union FragB16 { v16bf v; unsigned int u[8]; };
union Acc     { v8f   v; float        f[8]; };

// hardware RNE convert (v_cvt_*bf16) instead of software rounding
__device__ __forceinline__ unsigned short f2bf(float f) {
    __bf16 h = (__bf16)f;
    return __builtin_bit_cast(unsigned short, h);
}
__device__ __forceinline__ float bf2f(unsigned short h) {
    unsigned int u = ((unsigned int)h) << 16;
    return __builtin_bit_cast(float, u);
}
__device__ __forceinline__ unsigned int pack2(float a, float b) {
    union { __bf16 h[2]; unsigned int u; } p;
    p.h[0] = (__bf16)a;
    p.h[1] = (__bf16)b;
    return p.u;
}
// A-frag (16xK 16-bit): word v holds K = 2v + 16*hi
__device__ __forceinline__ int amap(int v, int hi) {
    return 2 * v + hi * 16;
}
// CDNA5 async global->LDS copy, 16B per lane (ISA §10.7 / §15.18.3)
__device__ __forceinline__ void async_copy_b128(unsigned int lds_off, const void* g) {
    unsigned long long ga = (unsigned long long)(size_t)g;
    asm volatile("global_load_async_to_lds_b128 %0, %1, off"
                 :: "v"(lds_off), "v"(ga) : "memory");
}
__device__ __forceinline__ void wait_asynccnt0() {
    asm volatile("s_wait_asynccnt 0x0" ::: "memory");
}

// ---------------------------------------------------------------------------
// K1: fused QKV projection.  z selects {Q,K,V}.  Out: bf16 [B,H,T,d]
// ---------------------------------------------------------------------------
__global__ void __launch_bounds__(128)
qkv_proj_kernel(const float* __restrict__ x, const float* __restrict__ ctx,
                const float* __restrict__ Wq, const float* __restrict__ Wk,
                const float* __restrict__ Wv,
                unsigned short* __restrict__ Qbf, unsigned short* __restrict__ Kbf,
                unsigned short* __restrict__ Vbf)
{
    __shared__ alignas(16) unsigned short As[64][32];   // [m][k] bf16
    __shared__ alignas(16) unsigned short Bs[64][32];   // [n][k] bf16 (transposed)
    const int z = blockIdx.z;
    const float* A = (z == 0) ? x : ctx;
    const float* W = (z == 0) ? Wq : ((z == 1) ? Wk : Wv);
    unsigned short* Out = (z == 0) ? Qbf : ((z == 1) ? Kbf : Vbf);
    const int m0 = blockIdx.x * 64, n0 = blockIdx.y * 64;
    const int tid = threadIdx.x, lane = tid & 31, wv = tid >> 5;
    const int lr = lane & 15, hi = lane >> 4;

    Acc acc[4];
    for (int t = 0; t < 4; ++t)
        for (int e = 0; e < 8; ++e) acc[t].f[e] = 0.f;

    for (int kk = 0; kk < NC; kk += 32) {
        __syncthreads();
        #pragma unroll
        for (int i = 0; i < 8; ++i) {                  // A: 64x32, float2 -> packed b32
            int idx = tid + 128 * i;                   // 0..1023 (pair granules)
            int r = idx >> 4, cp = idx & 15;
            float2 v2 = *(const float2*)(A + (size_t)(m0 + r) * NC + kk + cp * 2);
            *(unsigned int*)&As[r][cp * 2] = pack2(v2.x, v2.y);
        }
        #pragma unroll
        for (int i = 0; i < 8; ++i) {                  // W: 32x64 -> Bs transposed
            int idx = tid + 128 * i;                   // 0..1023
            int r = idx >> 5, c = (idx & 31) * 2;
            float2 v2 = *(const float2*)(W + (size_t)(kk + r) * NC + n0 + c);
            Bs[c][r]     = f2bf(v2.x);
            Bs[c + 1][r] = f2bf(v2.y);
        }
        __syncthreads();

        FragB16 a, bfr[4];
        const int am = wv * 16 + lr;
        #pragma unroll
        for (int v = 0; v < 8; ++v)
            a.u[v] = *(const unsigned int*)&As[am][amap(v, hi)];
        #pragma unroll
        for (int nt = 0; nt < 4; ++nt) {
            const int bn = nt * 16 + lr;
            #pragma unroll
            for (int v = 0; v < 8; ++v)
                bfr[nt].u[v] = *(const unsigned int*)&Bs[bn][2 * v + hi * 16];
        }
        #pragma unroll
        for (int nt = 0; nt < 4; ++nt)
            acc[nt].v = __builtin_amdgcn_wmma_f32_16x16x32_bf16(
                false, a.v, false, bfr[nt].v, (short)0, acc[nt].v, false, false);
    }
    // store: row = m0 + 16*wv + 8*hi + r ; col = n0 + 16*nt + lr
    const int b = m0 >> 10;
    const int tbase = (m0 & 1023) + wv * 16 + hi * 8;
    #pragma unroll
    for (int nt = 0; nt < 4; ++nt) {
        int col = n0 + nt * 16 + lr;
        int h = col >> 6, dd = col & 63;
        unsigned short* op = Out + (((size_t)b * NH + h) * NT + tbase) * ND + dd;
        #pragma unroll
        for (int r = 0; r < 8; ++r)
            op[(size_t)r * ND] = f2bf(acc[nt].f[r]);
    }
}

// ---------------------------------------------------------------------------
// K2: decomposed relative-position tables.
// rel_h[bh,t,kh] = sum_c Q[bh,t,c] * rel_pos_h[qh-kh+31, c]   (qh = t/32)
// ---------------------------------------------------------------------------
__global__ void __launch_bounds__(256)
relpos_kernel(const unsigned short* __restrict__ Qbf,
              const float* __restrict__ rph, const float* __restrict__ rpw,
              float* __restrict__ relH, float* __restrict__ relW)
{
    const int kidx = threadIdx.x;                       // 0..31
    const int t = blockIdx.x * 8 + threadIdx.y;         // 0..1023
    const int bh = blockIdx.y;
    const unsigned short* qrow = Qbf + ((size_t)bh * NT + t) * ND;
    const int qh = t >> 5, qw = t & 31;
    const float* th = rph + (size_t)(qh - kidx + 31) * ND;
    const float* tw = rpw + (size_t)(qw - kidx + 31) * ND;
    float sh = 0.f, sw = 0.f;
    #pragma unroll 8
    for (int c = 0; c < ND; ++c) {
        float qv = bf2f(qrow[c]);
        sh += qv * th[c];
        sw += qv * tw[c];
    }
    relH[((size_t)bh * NT + t) * 32 + kidx] = sh;
    relW[((size_t)bh * NT + t) * 32 + kidx] = sw;
}

// ---------------------------------------------------------------------------
// K3: flash attention, transposed (S^T = K*Q^T, O^T = V^T*P^T) so the online
// softmax is lane-local per query column (one shfl_xor(16) pair reduce).
// K tile staged with CDNA5 async global->LDS DMA; V transposed on stage.
// ---------------------------------------------------------------------------
__global__ void __launch_bounds__(128)
attn_kernel(const unsigned short* __restrict__ Qbf,
            const unsigned short* __restrict__ Kbf,
            const unsigned short* __restrict__ Vbf,
            const float* __restrict__ relH, const float* __restrict__ relW,
            unsigned short* __restrict__ Obf)
{
    __shared__ alignas(16) unsigned short Ks[32][64];   // [k][c]
    __shared__ alignas(16) unsigned short Vt[64][32];   // [d][k]  (V transposed)
    const int bh = blockIdx.y;
    const int b = bh / NH, h = bh % NH;
    const int tid = threadIdx.x, lane = tid & 31, wv = tid >> 5;
    const int lr = lane & 15, hi = lane >> 4;
    const int q = blockIdx.x * 64 + wv * 16 + lr;       // this lane's query column
    const size_t qoff = ((size_t)bh * NT + q) * ND;
    const unsigned int ks_base = (unsigned int)(size_t)&Ks[0][0];

    // Q^T B-fragments (c-dim split in two 32-chunks), live for whole kernel
    FragB16 bq[2];
    #pragma unroll
    for (int cc = 0; cc < 2; ++cc)
        #pragma unroll
        for (int v = 0; v < 8; ++v)
            bq[cc].u[v] = *(const unsigned int*)(Qbf + qoff + cc * 32 + 2 * v + hi * 16);

    // rel_w row repeats every 32 keys -> keep the 16 values this lane needs
    float rw[2][8];
    const float* rwrow = relW + ((size_t)bh * NT + q) * 32;
    #pragma unroll
    for (int j = 0; j < 2; ++j)
        #pragma unroll
        for (int r = 0; r < 8; ++r)
            rw[j][r] = rwrow[j * 16 + hi * 8 + r];
    const float* rhrow = relH + ((size_t)bh * NT + q) * 32;

    Acc o[4];
    for (int t = 0; t < 4; ++t)
        for (int e = 0; e < 8; ++e) o[t].f[e] = 0.f;
    float mstat = -3.0e38f, lstat = 0.f;

    for (int kb = 0; kb < NT; kb += 32) {
        __syncthreads();
        const unsigned short* kp = Kbf + ((size_t)bh * NT + kb) * ND;
        const unsigned short* vp = Vbf + ((size_t)bh * NT + kb) * ND;
        // K tile (4KB, layout-preserving): async DMA straight into LDS
        #pragma unroll
        for (int i = 0; i < 2; ++i) {
            int idx = tid + 128 * i;                    // 0..255 x 16B
            async_copy_b128(ks_base + (unsigned)idx * 16, kp + idx * 8);
        }
        // V tile: load u32 pairs, store transposed
        #pragma unroll
        for (int i = 0; i < 8; ++i) {
            int idx = tid + 128 * i;                    // u32 index 0..1023
            int r = idx >> 5, cp = idx & 31;
            unsigned int vd = *(const unsigned int*)(vp + (size_t)r * ND + cp * 2);
            Vt[cp * 2][r]     = (unsigned short)(vd & 0xFFFFu);
            Vt[cp * 2 + 1][r] = (unsigned short)(vd >> 16);
        }
        wait_asynccnt0();
        __syncthreads();

        // scores S^T: two 16k x 16q tiles;  abs key = kb + 16j + 8*hi + r
        FragB16 ak[2][2];
        #pragma unroll
        for (int j = 0; j < 2; ++j) {
            const int km = j * 16 + lr;
            #pragma unroll
            for (int cc = 0; cc < 2; ++cc)
                #pragma unroll
                for (int v = 0; v < 8; ++v)
                    ak[j][cc].u[v] = *(const unsigned int*)&Ks[km][cc * 32 + amap(v, hi)];
        }
        Acc st[2];
        #pragma unroll
        for (int j = 0; j < 2; ++j)
            for (int e = 0; e < 8; ++e) st[j].f[e] = 0.f;
        #pragma unroll
        for (int j = 0; j < 2; ++j) {
            st[j].v = __builtin_amdgcn_wmma_f32_16x16x32_bf16(
                false, ak[j][0].v, false, bq[0].v, (short)0, st[j].v, false, false);
            st[j].v = __builtin_amdgcn_wmma_f32_16x16x32_bf16(
                false, ak[j][1].v, false, bq[1].v, (short)0, st[j].v, false, false);
        }

        float pv[2][8];
        float localmax = -3.0e38f;
        const float rh = rhrow[kb >> 5];                // kh constant this iter
        #pragma unroll
        for (int j = 0; j < 2; ++j)
            #pragma unroll
            for (int r = 0; r < 8; ++r) {
                float val = st[j].f[r] * SCALE + rh + rw[j][r];
                pv[j][r] = val;
                localmax = fmaxf(localmax, val);
            }

        // online softmax (column stats shared between lane pair L, L+16)
        float om = fmaxf(localmax, __shfl_xor(localmax, 16, 32));
        float mnew = fmaxf(mstat, om);
        float alpha = __expf(mstat - mnew);
        float lsum = 0.f;
        unsigned int own32[2][4];
        #pragma unroll
        for (int j = 0; j < 2; ++j) {
            #pragma unroll
            for (int r = 0; r < 8; ++r) {
                float p = __expf(pv[j][r] - mnew);
                pv[j][r] = p;
                lsum += p;
            }
            #pragma unroll
            for (int w = 0; w < 4; ++w)
                own32[j][w] = pack2(pv[j][2 * w], pv[j][2 * w + 1]);
        }
        lsum += __shfl_xor(lsum, 16, 32);
        lstat = lstat * alpha + lsum;
        mstat = mnew;

        // assemble P^T B-fragment: lo lanes carry k=0..15 (tile0), hi k=16..31
        unsigned int par0[4], par1[4];
        #pragma unroll
        for (int w = 0; w < 4; ++w) {
            par0[w] = __shfl_xor(own32[0][w], 16, 32);
            par1[w] = __shfl_xor(own32[1][w], 16, 32);
        }
        FragB16 bp;
        #pragma unroll
        for (int w = 0; w < 4; ++w) {
            bp.u[w]     = hi ? par1[w]     : own32[0][w];
            bp.u[w + 4] = hi ? own32[1][w] : par0[w];
        }
        // O^T = O^T*alpha + V^T x P^T : scale, load all V frags, 4 WMMAs
        #pragma unroll
        for (int dd = 0; dd < 4; ++dd)
            #pragma unroll
            for (int e = 0; e < 8; ++e) o[dd].f[e] *= alpha;
        FragB16 av[4];
        #pragma unroll
        for (int dd = 0; dd < 4; ++dd) {
            const int dm = dd * 16 + lr;
            #pragma unroll
            for (int v = 0; v < 8; ++v)
                av[dd].u[v] = *(const unsigned int*)&Vt[dm][amap(v, hi)];
        }
        #pragma unroll
        for (int dd = 0; dd < 4; ++dd)
            o[dd].v = __builtin_amdgcn_wmma_f32_16x16x32_bf16(
                false, av[dd].v, false, bp.v, (short)0, o[dd].v, false, false);
    }

    const float inv = 1.0f / lstat;
    unsigned short* outp = Obf + ((size_t)b * NT + q) * NC + h * ND;
    #pragma unroll
    for (int dd = 0; dd < 4; ++dd)
        #pragma unroll
        for (int r = 0; r < 8; r += 2) {
            int d = dd * 16 + hi * 8 + r;               // even -> aligned u32 store
            *(unsigned int*)(outp + d) = pack2(o[dd].f[r] * inv, o[dd].f[r + 1] * inv);
        }
}

// ---------------------------------------------------------------------------
// K4: output projection  Y = Obf @ Wo + bo  (f32 out).  A tile is a pure bf16
// copy -> async global->LDS DMA overlapped with Wo convert staging.
// ---------------------------------------------------------------------------
__global__ void __launch_bounds__(128)
out_proj_kernel(const unsigned short* __restrict__ Obf,
                const float* __restrict__ Wo, const float* __restrict__ bo,
                float* __restrict__ Y)
{
    __shared__ alignas(16) unsigned short As[64][32];
    __shared__ alignas(16) unsigned short Bs[64][32];
    const int m0 = blockIdx.x * 64, n0 = blockIdx.y * 64;
    const int tid = threadIdx.x, lane = tid & 31, wv = tid >> 5;
    const int lr = lane & 15, hi = lane >> 4;
    const unsigned int as_base = (unsigned int)(size_t)&As[0][0];

    Acc acc[4];
    for (int t = 0; t < 4; ++t)
        for (int e = 0; e < 8; ++e) acc[t].f[e] = 0.f;

    for (int kk = 0; kk < NC; kk += 32) {
        __syncthreads();
        // A (bf16, layout-preserving within each 64B row chunk): async DMA
        #pragma unroll
        for (int i = 0; i < 2; ++i) {
            int idx = tid + 128 * i;                    // 0..255 x 16B
            int r = idx >> 2, cp = idx & 3;
            async_copy_b128(as_base + (unsigned)idx * 16,
                            Obf + (size_t)(m0 + r) * NC + kk + cp * 8);
        }
        #pragma unroll
        for (int i = 0; i < 8; ++i) {                   // Wo: 32x64 -> Bs transposed
            int idx = tid + 128 * i;
            int r = idx >> 5, c = (idx & 31) * 2;
            float2 v2 = *(const float2*)(Wo + (size_t)(kk + r) * NC + n0 + c);
            Bs[c][r]     = f2bf(v2.x);
            Bs[c + 1][r] = f2bf(v2.y);
        }
        wait_asynccnt0();
        __syncthreads();

        FragB16 a, bfr[4];
        const int am = wv * 16 + lr;
        #pragma unroll
        for (int v = 0; v < 8; ++v)
            a.u[v] = *(const unsigned int*)&As[am][amap(v, hi)];
        #pragma unroll
        for (int nt = 0; nt < 4; ++nt) {
            const int bn = nt * 16 + lr;
            #pragma unroll
            for (int v = 0; v < 8; ++v)
                bfr[nt].u[v] = *(const unsigned int*)&Bs[bn][2 * v + hi * 16];
        }
        #pragma unroll
        for (int nt = 0; nt < 4; ++nt)
            acc[nt].v = __builtin_amdgcn_wmma_f32_16x16x32_bf16(
                false, a.v, false, bfr[nt].v, (short)0, acc[nt].v, false, false);
    }
    const int rbase = m0 + wv * 16 + hi * 8;
    #pragma unroll
    for (int nt = 0; nt < 4; ++nt) {
        int col = n0 + nt * 16 + lr;
        float bias = bo[col];
        float* yp = Y + (size_t)rbase * NC + col;
        #pragma unroll
        for (int r = 0; r < 8; ++r)
            yp[(size_t)r * NC] = acc[nt].f[r] + bias;
    }
}

// ---------------------------------------------------------------------------
extern "C" void kernel_launch(void* const* d_in, const int* in_sizes, int n_in,
                              void* d_out, int out_size, void* d_ws, size_t ws_size,
                              hipStream_t stream)
{
    const float* x   = (const float*)d_in[0];
    const float* ctx = (const float*)d_in[1];
    const float* Wq  = (const float*)d_in[2];
    const float* Wk  = (const float*)d_in[3];
    const float* Wv  = (const float*)d_in[4];
    const float* Wo  = (const float*)d_in[5];
    const float* bo  = (const float*)d_in[6];
    const float* rph = (const float*)d_in[7];
    const float* rpw = (const float*)d_in[8];

    char* ws = (char*)d_ws;
    const size_t SEG = (size_t)NB * NH * NT * ND * 2;   // 12,582,912 B
    unsigned short* Qbf = (unsigned short*)(ws);
    unsigned short* Kbf = (unsigned short*)(ws + SEG);
    unsigned short* Vbf = (unsigned short*)(ws + 2 * SEG);
    float* relH         = (float*)(ws + 3 * SEG);
    float* relW         = (float*)(ws + 4 * SEG);
    unsigned short* Obf = (unsigned short*)(ws + 5 * SEG);

    qkv_proj_kernel<<<dim3(128, 12, 3), 128, 0, stream>>>(x, ctx, Wq, Wk, Wv,
                                                          Qbf, Kbf, Vbf);
    relpos_kernel<<<dim3(128, BH), dim3(32, 8), 0, stream>>>(Qbf, rph, rpw,
                                                             relH, relW);
    attn_kernel<<<dim3(16, BH), 128, 0, stream>>>(Qbf, Kbf, Vbf, relH, relW, Obf);
    out_proj_kernel<<<dim3(128, 12), 128, 0, stream>>>(Obf, Wo, bo, (float*)d_out);
}
